// RWKV_13692355739859
// MI455X (gfx1250) — matrix operand order repaired and checked by reading, not compiled, not run on hardware
//
#include <hip/hip_runtime.h>
#include <hip/hip_bf16.h>
#include <math.h>
#include <stdint.h>

// ---------------- problem dims ----------------
#define V_DIM 50304
#define L_DIM 12
#define C_DIM 768
#define FF_DIM 3072
#define B_DIM 2
#define T_DIM 1024
#define NROWS (B_DIM * T_DIM)        // 2048
#define NTC   (NROWS * C_DIM)        // 1,572,864
#define NTF   (NROWS * FF_DIM)       // 6,291,456
#define EPSV 1e-5f

// ---------------- WMMA types (CDNA5 / gfx1250, wave32) ----------------
typedef __attribute__((ext_vector_type(16))) __bf16 v16bf;
typedef __attribute__((ext_vector_type(8)))  __bf16 v8bf;
typedef __attribute__((ext_vector_type(8)))  float  v8f;

enum { EP_NONE = 0, EP_SIGMOID = 1, EP_RELUSQ = 2, EP_ADD = 3 };

// ======================================================================
// Embedding gather: x[b,t,:] = wte[idx[b,t],:]
// ======================================================================
__global__ void rwkv_embed(const int* __restrict__ idx,
                           const float* __restrict__ wte,
                           float* __restrict__ x) {
  int i = blockIdx.x * blockDim.x + threadIdx.x;
  if (i >= NTC) return;
  int bt = i / C_DIM;
  int c  = i - bt * C_DIM;
  x[i] = wte[(size_t)idx[bt] * C_DIM + c];
}

// ======================================================================
// LayerNorm over C=768: one wave32 per row, 24 elements per lane.
// ======================================================================
__global__ void rwkv_ln(const float* __restrict__ x,
                        const float* __restrict__ w,
                        const float* __restrict__ b,
                        float* __restrict__ out) {
  int wid  = (blockIdx.x * blockDim.x + threadIdx.x) >> 5;
  int lane = threadIdx.x & 31;
  if (wid >= NROWS) return;
  const float* xr = x + (size_t)wid * C_DIM;
  float v[C_DIM / 32];
  float s = 0.f;
#pragma unroll
  for (int i = 0; i < C_DIM / 32; ++i) { v[i] = xr[lane + i * 32]; s += v[i]; }
#pragma unroll
  for (int o = 16; o > 0; o >>= 1) s += __shfl_xor(s, o, 32);
  float mu = s * (1.0f / C_DIM);
  float q = 0.f;
#pragma unroll
  for (int i = 0; i < C_DIM / 32; ++i) { float d = v[i] - mu; q += d * d; }
#pragma unroll
  for (int o = 16; o > 0; o >>= 1) q += __shfl_xor(q, o, 32);
  float inv = rsqrtf(q * (1.0f / C_DIM) + EPSV);
  float* op = out + (size_t)wid * C_DIM;
#pragma unroll
  for (int i = 0; i < C_DIM / 32; ++i) {
    int c = lane + i * 32;
    op[c] = (v[i] - mu) * inv * w[c] + b[c];
  }
}

// ======================================================================
// time-shift + token mix: xr = h*mr + prev(h)*(1-mr), etc. (xv optional)
// ======================================================================
__global__ void rwkv_shift_mix(const float* __restrict__ h,
                               const float* __restrict__ mr,
                               const float* __restrict__ mk,
                               const float* __restrict__ mv,
                               float* __restrict__ xr,
                               float* __restrict__ xk,
                               float* __restrict__ xv) {
  int i = blockIdx.x * blockDim.x + threadIdx.x;
  if (i >= NTC) return;
  int c = i % C_DIM;
  int t = (i / C_DIM) % T_DIM;
  float hv = h[i];
  float pv = (t > 0) ? h[i - C_DIM] : 0.f;
  float a = mr[c]; xr[i] = hv * a + pv * (1.f - a);
  a = mk[c];       xk[i] = hv * a + pv * (1.f - a);
  if (xv != nullptr) { a = mv[c]; xv[i] = hv * a + pv * (1.f - a); }
}

// ======================================================================
// WKV scan: one lane per (b,c) channel, serial over T (stabilized form).
// K/V are streamed into LDS with CDNA5 async-LDS loads (ASYNCcnt),
// double-buffered: chunk i+1 streams while chunk i's serial steps run.
// Each lane only reads the LDS words loaded by its own wave, so
// s_wait_asynccnt alone orders consumption (no block barrier needed).
// ======================================================================
#define WTS 8                         // timesteps per chunk
#define WKV_ISSUE(T0, BUF)                                                     \
  _Pragma("unroll")                                                            \
  for (int tt = 0; tt < WTS; ++tt) {                                           \
    unsigned dk = (unsigned)(uintptr_t)&lk[BUF][tt][threadIdx.x];              \
    unsigned dv = (unsigned)(uintptr_t)&lv[BUF][tt][threadIdx.x];              \
    unsigned long long gk =                                                    \
        (unsigned long long)(Kp + base + (size_t)((T0) + tt) * C_DIM);         \
    unsigned long long gv =                                                    \
        (unsigned long long)(Vp + base + (size_t)((T0) + tt) * C_DIM);         \
    asm volatile("global_load_async_to_lds_b32 %0, %1, off"                    \
                 ::"v"(dk), "v"(gk) : "memory");                               \
    asm volatile("global_load_async_to_lds_b32 %0, %1, off"                    \
                 ::"v"(dv), "v"(gv) : "memory");                               \
  }

__global__ __launch_bounds__(256)
void rwkv_wkv(const float* __restrict__ Kp, const float* __restrict__ Vp,
              const float* __restrict__ td, const float* __restrict__ tf,
              float* __restrict__ out) {
  __shared__ float lk[2][WTS][256];
  __shared__ float lv[2][WTS][256];
  const int nblkc = C_DIM / 256;                       // 3 blocks per batch
  const int b = blockIdx.x / nblkc;
  const int c = (blockIdx.x % nblkc) * 256 + threadIdx.x;
  const float w = -expf(td[c]);
  const float u = tf[c];
  float num = 0.f, den = 0.f, mx = -1e38f;
  const size_t base = (size_t)b * T_DIM * C_DIM + c;

  WKV_ISSUE(0, 0)                                      // prefetch chunk 0
  int cur = 0;
  for (int t0 = 0; t0 < T_DIM; t0 += WTS) {
    const bool more = (t0 + WTS) < T_DIM;
    if (more) {
      WKV_ISSUE(t0 + WTS, cur ^ 1)                     // stream next chunk
      // 2*WTS=16 ops outstanding for next chunk; async loads complete in
      // order, so <=16 outstanding  =>  current chunk is resident.
      asm volatile("s_wait_asynccnt 0x10" ::: "memory");
    } else {
      asm volatile("s_wait_asynccnt 0x0" ::: "memory");
    }
#pragma unroll
    for (int t = 0; t < WTS; ++t) {
      float kk = ((volatile const float*)&lk[cur][t][threadIdx.x])[0];
      float vv = ((volatile const float*)&lv[cur][t][threadIdx.x])[0];
      float mfo = fmaxf(mx, kk + u);
      float e1  = expf(mx - mfo);
      float e2  = expf(kk + u - mfo);
      out[base + (size_t)(t0 + t) * C_DIM] = (e1 * num + e2 * vv) / (e1 * den + e2);
      float mfs = fmaxf(mx + w, kk);
      float e1s = expf(mx + w - mfs);
      float e2s = expf(kk - mfs);
      num = e1s * num + e2s * vv;
      den = e1s * den + e2s;
      mx  = mfs;
    }
    cur ^= 1;
  }
}

// ======================================================================
// elementwise helpers
// ======================================================================
__global__ void rwkv_mul(const float* __restrict__ a, const float* __restrict__ b,
                         float* __restrict__ o) {
  int i = blockIdx.x * blockDim.x + threadIdx.x;
  if (i < NTC) o[i] = a[i] * b[i];
}

__global__ void rwkv_addmul(float* __restrict__ x, const float* __restrict__ a,
                            const float* __restrict__ b) {
  int i = blockIdx.x * blockDim.x + threadIdx.x;
  if (i < NTC) x[i] = x[i] + a[i] * b[i];
}

// ======================================================================
// bf16 WMMA GEMM:  C[M,N] (f32) = epilogue( A[M,K] (f32) @ W[N,K]^T )
//   Block tile 128x128xK32, 256 threads = 8 wave32s (4x2 wave grid),
//   each wave computes 32x64 = 2x4 v_wmma_f32_16x16x32_bf16 tiles.
//   Register double-buffer (global) + LDS ping-pong: one barrier/iter.
//   All launch shapes divide the tiles exactly -> no bounds guards.
// ======================================================================
#define BM 128
#define BN 128
#define BK 32
#define LDF 40   // LDS row stride in halves: 80B, keeps 16B alignment

// f32x8 -> bf16x8 in one convertvector: lowers to 4x v_cvt_pk_bf16_f32,
// bit-cast to uint4 for a single ds_store_b128 (no half-splicing VALU).
__device__ __forceinline__ uint4 pack8(float4 a, float4 b) {
  v8f f;
  f[0] = a.x; f[1] = a.y; f[2] = a.z; f[3] = a.w;
  f[4] = b.x; f[5] = b.y; f[6] = b.z; f[7] = b.w;
  v8bf bb = __builtin_convertvector(f, v8bf);
  return __builtin_bit_cast(uint4, bb);
}

__global__ __launch_bounds__(256)
void rwkv_gemm_bf16(const float* __restrict__ A, const float* __restrict__ W,
                    float* __restrict__ Cm, int M, int N, int K, int mode) {
  __shared__ __bf16 As[2][BM * LDF];
  __shared__ __bf16 Bs[2][BN * LDF];

  const int tid  = threadIdx.x;
  const int lane = tid & 31;
  const int wave = tid >> 5;
  const int wm   = wave & 3;   // 4 waves along M (32 rows each)
  const int wn   = wave >> 2;  // 2 waves along N (64 cols each)
  const int bm   = blockIdx.y * BM;
  const int bn   = blockIdx.x * BN;

  v8f acc[2][4];
#pragma unroll
  for (int i = 0; i < 2; ++i)
#pragma unroll
    for (int j = 0; j < 4; ++j)
#pragma unroll
      for (int e = 0; e < 8; ++e) acc[i][j][e] = 0.f;

  // tile loader: 256 threads cover 128 rows x 32 cols, 16 f32 each
  const int lr = tid >> 1;
  const int lc = (tid & 1) * 16;
  const float* ap = A + (size_t)(bm + lr) * K + lc;
  const float* wp = W + (size_t)(bn + lr) * K + lc;

  // fragment geometry (wave32 WMMA layouts, ISA 7.12.2)
  const int fr  = lane & 15;         // row within 16 (M for A, N for B)
  const int fhA = (lane >> 4) << 3;  // A: K-halves {h..h+7, h+16..h+23}
  const int fkB = (lane >> 4) << 3;  // B: K-halves {h..h+7, h+16..h+23}

  float4 ra[4], rw[4];
#pragma unroll
  for (int u = 0; u < 4; ++u) {
    ra[u] = *(const float4*)(ap + 4 * u);
    rw[u] = *(const float4*)(wp + 4 * u);
  }
  {
    __bf16* as_ = &As[0][lr * LDF + lc];
    __bf16* bs_ = &Bs[0][lr * LDF + lc];
    *(uint4*)(as_)     = pack8(ra[0], ra[1]);
    *(uint4*)(as_ + 8) = pack8(ra[2], ra[3]);
    *(uint4*)(bs_)     = pack8(rw[0], rw[1]);
    *(uint4*)(bs_ + 8) = pack8(rw[2], rw[3]);
  }
  __syncthreads();

  int cur = 0;
  for (int k0 = 0; k0 < K; k0 += BK) {
    const bool more = (k0 + BK) < K;
    if (more) {
#pragma unroll
      for (int u = 0; u < 4; ++u) {
        ra[u] = *(const float4*)(ap + k0 + BK + 4 * u);
        rw[u] = *(const float4*)(wp + k0 + BK + 4 * u);
      }
    }

    v16bf af[2];
#pragma unroll
    for (int i = 0; i < 2; ++i) {
      const __bf16* p = &As[cur][(wm * 32 + i * 16 + fr) * LDF + fhA];
      v8bf lo = *(const v8bf*)(p);
      v8bf hi = *(const v8bf*)(p + 16);
      af[i] = __builtin_shufflevector(lo, hi, 0, 1, 2, 3, 4, 5, 6, 7,
                                      8, 9, 10, 11, 12, 13, 14, 15);
    }
    v16bf bfr[4];
#pragma unroll
    for (int j = 0; j < 4; ++j) {
      const __bf16* p = &Bs[cur][(wn * 64 + j * 16 + fr) * LDF + fkB];
      v8bf lo = *(const v8bf*)(p);
      v8bf hi = *(const v8bf*)(p + 16);
      bfr[j] = __builtin_shufflevector(lo, hi, 0, 1, 2, 3, 4, 5, 6, 7,
                                       8, 9, 10, 11, 12, 13, 14, 15);
    }
#pragma unroll
    for (int i = 0; i < 2; ++i)
#pragma unroll
      for (int j = 0; j < 4; ++j)
        acc[i][j] = __builtin_amdgcn_wmma_f32_16x16x32_bf16(
            false, af[i], false, bfr[j], (short)0, acc[i][j], false, false);

    if (more) {
      __bf16* as_ = &As[cur ^ 1][lr * LDF + lc];
      __bf16* bs_ = &Bs[cur ^ 1][lr * LDF + lc];
      *(uint4*)(as_)     = pack8(ra[0], ra[1]);
      *(uint4*)(as_ + 8) = pack8(ra[2], ra[3]);
      *(uint4*)(bs_)     = pack8(rw[0], rw[1]);
      *(uint4*)(bs_ + 8) = pack8(rw[2], rw[3]);
    }
    __syncthreads();
    cur ^= 1;
  }

  // epilogue: D layout — VGPR v holds M = v + 8*(lane>=16), N = lane%16
  const int moff = (lane >> 4) * 8;
  const int nl   = lane & 15;
#pragma unroll
  for (int i = 0; i < 2; ++i)
#pragma unroll
    for (int j = 0; j < 4; ++j) {
      const int gn = bn + wn * 64 + j * 16 + nl;
#pragma unroll
      for (int vv = 0; vv < 8; ++vv) {
        const int gm = bm + wm * 32 + i * 16 + moff + vv;
        float val = acc[i][j][vv];
        size_t o = (size_t)gm * N + gn;
        if (mode == EP_SIGMOID)      val = 1.f / (1.f + expf(-val));
        else if (mode == EP_RELUSQ)  { val = fmaxf(val, 0.f); val = val * val; }
        else if (mode == EP_ADD)     val += Cm[o];
        Cm[o] = val;
      }
    }
}

static inline void launch_gemm(const float* A, const float* W, float* C,
                               int M, int N, int K, int mode, hipStream_t s) {
  dim3 g(N / BN, M / BM);   // all shapes divide exactly
  rwkv_gemm_bf16<<<g, 256, 0, s>>>(A, W, C, M, N, K, mode);
}

// ======================================================================
// host orchestration
// ======================================================================
extern "C" void kernel_launch(void* const* d_in, const int* in_sizes, int n_in,
                              void* d_out, int out_size, void* d_ws, size_t ws_size,
                              hipStream_t stream) {
  const int*   idx   = (const int*)d_in[0];
  const float* wte   = (const float*)d_in[1];
  const float* lnpw  = (const float*)d_in[2];
  const float* lnpb  = (const float*)d_in[3];
  const float* ln1w  = (const float*)d_in[4];
  const float* ln1b  = (const float*)d_in[5];
  const float* ln2w  = (const float*)d_in[6];
  const float* ln2b  = (const float*)d_in[7];
  const float* tmWk  = (const float*)d_in[8];
  const float* tmWv  = (const float*)d_in[9];
  const float* tmWr  = (const float*)d_in[10];
  const float* tmWo  = (const float*)d_in[11];
  const float* td    = (const float*)d_in[12];
  const float* tf    = (const float*)d_in[13];
  const float* mixk  = (const float*)d_in[14];
  const float* mixv  = (const float*)d_in[15];
  const float* mixr  = (const float*)d_in[16];
  const float* cmWk  = (const float*)d_in[17];
  const float* cmWv  = (const float*)d_in[18];
  const float* cmWr  = (const float*)d_in[19];
  const float* cmixk = (const float*)d_in[20];
  const float* cmixr = (const float*)d_in[21];
  const float* lnfw  = (const float*)d_in[22];
  const float* lnfb  = (const float*)d_in[23];
  const float* lmW   = (const float*)d_in[24];
  float* out = (float*)d_out;

  // workspace layout (floats): 9 C-buffers + 1 FF-buffer + 1 C-buffer ~ 88 MB
  float* ws  = (float*)d_ws;
  float* x   = ws;
  float* h   = x   + NTC;
  float* xr  = h   + NTC;
  float* xk  = xr  + NTC;
  float* xv  = xk  + NTC;
  float* r   = xv  + NTC;
  float* kb  = r   + NTC;
  float* vb  = kb  + NTC;
  float* wkv = vb  + NTC;
  float* kk  = wkv + NTC;   // NTF elements
  float* v2  = kk  + NTF;

  const int EW = 256;
  dim3 eg((NTC + EW - 1) / EW);
  dim3 lng(NROWS / 8);      // 8 wave32s per block -> one row per wave

  rwkv_embed<<<eg, EW, 0, stream>>>(idx, wte, x);
  rwkv_ln<<<lng, 256, 0, stream>>>(x, lnpw, lnpb, x);

  for (int l = 0; l < L_DIM; ++l) {
    size_t oc  = (size_t)l * C_DIM;
    size_t occ = (size_t)l * C_DIM * C_DIM;
    // ---- time mixing ----
    rwkv_ln<<<lng, 256, 0, stream>>>(x, ln1w + oc, ln1b + oc, h);
    rwkv_shift_mix<<<eg, EW, 0, stream>>>(h, mixr + oc, mixk + oc, mixv + oc,
                                          xr, xk, xv);
    launch_gemm(xr, tmWr + occ, r,  NROWS, C_DIM, C_DIM, EP_SIGMOID, stream);
    launch_gemm(xk, tmWk + occ, kb, NROWS, C_DIM, C_DIM, EP_NONE, stream);
    launch_gemm(xv, tmWv + occ, vb, NROWS, C_DIM, C_DIM, EP_NONE, stream);
    rwkv_wkv<<<dim3(B_DIM * (C_DIM / 256)), 256, 0, stream>>>(
        kb, vb, td + oc, tf + oc, wkv);
    rwkv_mul<<<eg, EW, 0, stream>>>(r, wkv, xr);            // xr := r*wkv
    launch_gemm(xr, tmWo + occ, x, NROWS, C_DIM, C_DIM, EP_ADD, stream);
    // ---- channel mixing ----
    rwkv_ln<<<lng, 256, 0, stream>>>(x, ln2w + oc, ln2b + oc, h);
    rwkv_shift_mix<<<eg, EW, 0, stream>>>(h, cmixr + oc, cmixk + oc, nullptr,
                                          xr, xk, nullptr);
    launch_gemm(xr, cmWr + occ, r, NROWS, C_DIM, C_DIM, EP_SIGMOID, stream);
    launch_gemm(xk, cmWk + (size_t)l * FF_DIM * C_DIM, kk,
                NROWS, FF_DIM, C_DIM, EP_RELUSQ, stream);
    launch_gemm(kk, cmWv + (size_t)l * C_DIM * FF_DIM, v2,
                NROWS, C_DIM, FF_DIM, EP_NONE, stream);
    rwkv_addmul<<<eg, EW, 0, stream>>>(x, r, v2);           // x += r2*v2
  }

  rwkv_ln<<<lng, 256, 0, stream>>>(x, lnfw, lnfb, h);
  launch_gemm(h, lmW, out, NROWS, V_DIM, C_DIM, EP_NONE, stream);
}
